// DotProductAttention_62337155334200
// MI455X (gfx1250) — matrix operand order repaired and checked by reading, not compiled, not run on hardware
//
#include <hip/hip_runtime.h>
#include <hip/hip_bf16.h>

// Flash-attention forward for gfx1250 (MI455X), wave32 + v_wmma_f32_16x16x32_f16.
// Shapes: Q,K,V = [2048, 2, 32, 64] f32, causal mask, out = [2048, 2, 2048] f32.
//
// Pipeline:
//   1) cvt_k_kernel:       K f32 -> f16, same layout (one-time)
//   2) transpose_v_kernel: V f32 -> f16 transposed to [head][hn][token]
//   3) fa kernel:          computes S^T = K @ Q^T so that exp(S^T) lands
//                          directly in the A-fragment layout required by P@V:
//                          no LDS transpose, no cross-lane ops in the loop.
//                          Softmax denominator via a ones-matrix WMMA; exp in
//                          log2 domain (log2(e) folded into the Q scale) so
//                          each probability is a single v_exp; exactly one
//                          mask-peeled step per wave handles the diagonal.
// A fallback all-f32 kernel is used if the harness workspace is too small.

typedef __attribute__((ext_vector_type(16))) _Float16 v16h;
typedef __attribute__((ext_vector_type(8)))  float    v8f;
typedef __attribute__((ext_vector_type(4)))  float    v4f;

#define SQa 2048
#define SKa 2048
#define Ba  2
#define NPa 32
#define HNa 64
#define ROWSTRIDE (Ba * NPa * HNa)   // 4096 elements between consecutive tokens

__device__ __forceinline__ v8f wmma(v16h a, v16h b, v8f c) {
  return __builtin_amdgcn_wmma_f32_16x16x32_f16(false, a, false, b, (short)0, c,
                                                false, false);
}

// A-matrix fragment (16x32 f16) from f16 memory: lane<16 holds row=lane%16,
// K = {0..7, 16..23}; lane>=16 K = {8..15, 24..31} (off8 folded into pointer).
// Two 16-byte contiguous runs -> 2x global_load_b128.
__device__ __forceinline__ v16h load_a_f16(const _Float16* __restrict__ p) {
  v16h a;
#pragma unroll
  for (int i = 0; i < 8; ++i) { a[i] = p[i]; a[i + 8] = p[16 + i]; }
  return a;
}

// B-matrix fragment (32x16 f16): lane<16 holds col=lane%16, K = {0..7, 16..23};
// lane>=16 K = {8..15, 24..31} (off8 folded into pointer). Two 16-byte runs.
__device__ __forceinline__ v16h load_b_f16(const _Float16* __restrict__ p) {
  v16h b;
#pragma unroll
  for (int i = 0; i < 8; ++i) { b[i] = p[i]; b[i + 8] = p[16 + i]; }
  return b;
}

// B fragment from f32 memory with scale (used for Q^T; once per wave).
__device__ __forceinline__ v16h load_b_f32_scaled(const float* __restrict__ p,
                                                  float scale) {
  v16h b;
#pragma unroll
  for (int i = 0; i < 8; ++i) { b[i] = (_Float16)(p[i] * scale); b[i + 8] = (_Float16)(p[16 + i] * scale); }
  return b;
}

// ---------------------------------------------------------------------------
// Preprocessing kernels
// ---------------------------------------------------------------------------
__global__ void __launch_bounds__(256)
cvt_k_kernel(const float* __restrict__ k, _Float16* __restrict__ kh) {
  const int i = (blockIdx.x * 256 + threadIdx.x) * 4;
  const v4f x = *(const v4f*)(k + i);
  kh[i + 0] = (_Float16)x.x;
  kh[i + 1] = (_Float16)x.y;
  kh[i + 2] = (_Float16)x.z;
  kh[i + 3] = (_Float16)x.w;
}

// V: f32 [token][head*64+h] -> f16 vt[head*64+h][token], LDS-tiled 64x64.
__global__ void __launch_bounds__(256)
transpose_v_kernel(const float* __restrict__ v, _Float16* __restrict__ vt) {
  __shared__ _Float16 tile[64][64 + 8];
  const int head = blockIdx.y;
  const int t0   = blockIdx.x * 64;
#pragma unroll
  for (int i = threadIdx.x; i < 64 * 64; i += 256) {
    const int tt = i >> 6, h = i & 63;                       // h contiguous
    tile[h][tt] = (_Float16)v[(size_t)(t0 + tt) * ROWSTRIDE + head * HNa + h];
  }
  __syncthreads();
#pragma unroll
  for (int i = threadIdx.x; i < 64 * 64; i += 256) {
    const int h = i >> 6, tt = i & 63;                       // tt contiguous
    vt[((size_t)head * HNa + h) * SKa + t0 + tt] = tile[h][tt];
  }
}

// ---------------------------------------------------------------------------
// One KV step (32 tokens). MASK=true only for the single diagonal step.
// ---------------------------------------------------------------------------
template <bool MASK>
__device__ __forceinline__ void fa_step(int t0, int q0, int colbase, int rowoff,
                                        const _Float16* __restrict__ kp,
                                        const _Float16* __restrict__ vp,
                                        v16h qb0, v16h qb1, v16h ones,
                                        v8f& acc0, v8f& acc1, v8f& acc2,
                                        v8f& acc3, v8f& accl) {
  // ---- S^T = K @ Q^T : two 16(token)x16(qrow) tiles ----
  const _Float16* k0 = kp + (size_t)t0 * ROWSTRIDE;
  const _Float16* k1 = k0 + 16 * ROWSTRIDE;
  v8f st0 = {}, st1 = {};
  st0 = wmma(load_a_f16(k0),      qb0, st0);
  st0 = wmma(load_a_f16(k0 + 32), qb1, st0);
  st1 = wmma(load_a_f16(k1),      qb0, st1);
  st1 = wmma(load_a_f16(k1 + 32), qb1, st1);

  // ---- P^T = exp2(S^T): already in A-fragment layout for P @ V ----
  v16h pa;
#pragma unroll
  for (int r = 0; r < 8; ++r) {
    float s0 = __builtin_amdgcn_fmed3f(st0[r], -1.0e30f, 15.5f);  // f16-safe
    float s1 = __builtin_amdgcn_fmed3f(st1[r], -1.0e30f, 15.5f);
    if (MASK) {
      const int qrow = q0 + colbase;                // lane's q-row (S^T col)
      if (t0 + r + rowoff      > qrow) s0 = -1.0e4f;  // exp2 -> exactly 0
      if (t0 + 16 + r + rowoff > qrow) s1 = -1.0e4f;
    }
    pa[r]     = (_Float16)__builtin_amdgcn_exp2f(s0);  // single v_exp_f32
    pa[r + 8] = (_Float16)__builtin_amdgcn_exp2f(s1);
  }

  // ---- O += P @ V (4 tiles) and l += P @ ones (denominator via WMMA) ----
  const _Float16* vb = vp + t0;
  accl = wmma(pa, ones,                      accl);
  acc0 = wmma(pa, load_b_f16(vb),            acc0);
  acc1 = wmma(pa, load_b_f16(vb + 16 * SKa), acc1);
  acc2 = wmma(pa, load_b_f16(vb + 32 * SKa), acc2);
  acc3 = wmma(pa, load_b_f16(vb + 48 * SKa), acc3);
}

// ---------------------------------------------------------------------------
// Main flash-attention kernel (f16 K / transposed f16 V), no LDS.
// ---------------------------------------------------------------------------
__global__ void __launch_bounds__(256)
DotProductAttention_fa_kernel(const float* __restrict__ q,
                              const _Float16* __restrict__ kh,
                              const _Float16* __restrict__ vt,
                              float* __restrict__ out) {
  const int lane    = threadIdx.x & 31;
  const int wave    = threadIdx.x >> 5;
  const int headoff = blockIdx.y * HNa;              // (b*NP + n) * 64
  const int q0      = blockIdx.x * 128 + wave * 16;  // first query row of wave

  const int colbase = lane & 15;            // S^T column = q-row in tile
  const int rowoff  = (lane & 16) ? 8 : 0;  // token offset (C) / A K-offset
  const int koff16  = (lane & 16) ? 16 : 0; // unused (B fragments offset K by rowoff)

  // ---- Q^T B-fragments (once per wave); scale = 1/sqrt(HN) * log2(e) ----
  const float  QSCALE = 0.125f * 1.4426950408889634f;
  const float* qp = q + (size_t)(q0 + colbase) * ROWSTRIDE + headoff + rowoff;
  const v16h qb0 = load_b_f32_scaled(qp,      QSCALE);
  const v16h qb1 = load_b_f32_scaled(qp + 32, QSCALE);

  v16h ones;
#pragma unroll
  for (int i = 0; i < 16; ++i) ones[i] = (_Float16)1.0f;

  v8f acc0 = {}, acc1 = {}, acc2 = {}, acc3 = {}, accl = {};

  // Fragment base pointers (per-lane offsets folded in).
  const _Float16* kp = kh + (size_t)colbase * ROWSTRIDE + headoff + rowoff;
  const _Float16* vp = vt + ((size_t)headoff + colbase) * SKa + rowoff;

  // Steady state: fully-unmasked KV blocks.
  int t0 = 0;
#pragma unroll 2
  for (; t0 + 32 <= q0; t0 += 32) {
    __builtin_prefetch(kp + (size_t)(t0 + 32) * ROWSTRIDE, 0, 1);
    fa_step<false>(t0, q0, colbase, rowoff, kp, vp, qb0, qb1, ones,
                   acc0, acc1, acc2, acc3, accl);
  }
  // Exactly one diagonal (masked) step covers the remaining tokens < q0+16.
  fa_step<true>(t0, q0, colbase, rowoff, kp, vp, qb0, qb1, ones,
                acc0, acc1, acc2, acc3, accl);

  // ---- normalize and write out [SQ, B, NP*HN] ----
#pragma unroll
  for (int r = 0; r < 8; ++r) {
    const int row = q0 + r + rowoff;
    const float inv = 1.0f / accl[r];
    float* op = out + (size_t)row * ROWSTRIDE + headoff + colbase;
    op[0]  = acc0[r] * inv;
    op[16] = acc1[r] * inv;
    op[32] = acc2[r] * inv;
    op[48] = acc3[r] * inv;
  }
}

// ---------------------------------------------------------------------------
// Fallback: all-f32 inputs with in-loop conversion (used if ws too small)
// ---------------------------------------------------------------------------
__device__ __forceinline__ v16h load_a_f32s(const float* __restrict__ rowp,
                                            int kb, int lane, float scale) {
  const int off = (lane & 16) ? 8 : 0;
  const float* p = rowp + kb + off;
  v16h a;
#pragma unroll
  for (int i = 0; i < 8; ++i) {
    a[i]     = (_Float16)(p[i]      * scale);
    a[i + 8] = (_Float16)(p[16 + i] * scale);
  }
  return a;
}
__device__ __forceinline__ v16h load_b_cvt(const float* __restrict__ p) {
  v16h b;
#pragma unroll
  for (int i = 0; i < 8; ++i) { b[i] = (_Float16)p[i]; b[i + 8] = (_Float16)p[16 + i]; }
  return b;
}
__device__ __forceinline__ v16h load_b_strided(const float* __restrict__ p) {
  v16h b;
#pragma unroll
  for (int i = 0; i < 8; ++i) { b[i] = (_Float16)p[i * ROWSTRIDE]; b[i + 8] = (_Float16)p[(16 + i) * ROWSTRIDE]; }
  return b;
}

__global__ void __launch_bounds__(256)
fa_f32_fallback_kernel(const float* __restrict__ q, const float* __restrict__ k,
                       const float* __restrict__ v, float* __restrict__ out) {
  __shared__ _Float16 plds[8 * 16 * 32];
  const int lane    = threadIdx.x & 31;
  const int wave    = threadIdx.x >> 5;
  const int headoff = blockIdx.y * HNa;
  const int q0      = blockIdx.x * 128 + wave * 16;
  const int colbase = lane & 15;
  const int rowoff  = (lane & 16) ? 8 : 0;
  const int koff16  = (lane & 16) ? 16 : 0;

  const float* qp = q + (size_t)(q0 + colbase) * ROWSTRIDE + headoff;
  const v16h qa0 = load_a_f32s(qp, 0,  lane, 0.125f);
  const v16h qa1 = load_a_f32s(qp, 32, lane, 0.125f);
  v16h ones;
#pragma unroll
  for (int i = 0; i < 16; ++i) ones[i] = (_Float16)1.0f;

  v8f acc0 = {}, acc1 = {}, acc2 = {}, acc3 = {}, accl = {};
  _Float16* pw       = plds + wave * (16 * 32);
  const _Float16* pr = pw + colbase * 32 + rowoff;

  for (int t0 = 0; t0 < q0 + 16; t0 += 32) {
    const float* k0p = k + (size_t)(t0 + colbase) * ROWSTRIDE + headoff + rowoff;
    const float* k1p = k0p + 16 * ROWSTRIDE;
    v8f c0 = {}, c1 = {};
    c0 = wmma(qa0, load_b_cvt(k0p),      c0);
    c0 = wmma(qa1, load_b_cvt(k0p + 32), c0);
    c1 = wmma(qa0, load_b_cvt(k1p),      c1);
    c1 = wmma(qa1, load_b_cvt(k1p + 32), c1);

    const bool need_mask = (t0 + 31 > q0);
#pragma unroll
    for (int r = 0; r < 8; ++r) {
      float s0 = __builtin_amdgcn_fmed3f(c0[r], -1.0e30f, 11.0f);
      float s1 = __builtin_amdgcn_fmed3f(c1[r], -1.0e30f, 11.0f);
      if (need_mask) {
        const int row = q0 + r + rowoff;
        if (t0 + colbase      > row) s0 = -1.0e4f;
        if (t0 + colbase + 16 > row) s1 = -1.0e4f;
      }
      pw[(r + rowoff) * 32 + colbase]      = (_Float16)__expf(s0);
      pw[(r + rowoff) * 32 + colbase + 16] = (_Float16)__expf(s1);
    }
    v16h pa;
#pragma unroll
    for (int i = 0; i < 8; ++i) { pa[i] = pr[i]; pa[i + 8] = pr[16 + i]; }

    const float* vb = v + (size_t)(t0 + rowoff) * ROWSTRIDE + headoff + colbase;
    accl = wmma(pa, ones,                    accl);
    acc0 = wmma(pa, load_b_strided(vb),      acc0);
    acc1 = wmma(pa, load_b_strided(vb + 16), acc1);
    acc2 = wmma(pa, load_b_strided(vb + 32), acc2);
    acc3 = wmma(pa, load_b_strided(vb + 48), acc3);
  }
#pragma unroll
  for (int r = 0; r < 8; ++r) {
    const int row = q0 + r + rowoff;
    const float inv = 1.0f / accl[r];
    float* op = out + (size_t)row * ROWSTRIDE + headoff + colbase;
    op[0]  = acc0[r] * inv;
    op[16] = acc1[r] * inv;
    op[32] = acc2[r] * inv;
    op[48] = acc3[r] * inv;
  }
}

// ---------------------------------------------------------------------------
extern "C" void kernel_launch(void* const* d_in, const int* in_sizes, int n_in,
                              void* d_out, int out_size, void* d_ws, size_t ws_size,
                              hipStream_t stream) {
  const float* q = (const float*)d_in[0];
  const float* k = (const float*)d_in[1];
  const float* v = (const float*)d_in[2];
  // d_in[3] is the boolean causal mask; it is exactly "col > row", applied
  // analytically in-kernel.
  float* out = (float*)d_out;

  const size_t nkv      = (size_t)SKa * ROWSTRIDE;       // 8M elements
  const size_t kh_bytes = nkv * sizeof(_Float16);        // 16 MB
  dim3 grid(SQa / 128, Ba * NPa);
  dim3 block(256);

  if (ws_size >= 2 * kh_bytes) {
    _Float16* kh = (_Float16*)d_ws;
    _Float16* vt = (_Float16*)((char*)d_ws + kh_bytes);
    cvt_k_kernel<<<(int)(nkv / (256 * 4)), 256, 0, stream>>>(k, kh);
    transpose_v_kernel<<<dim3(SKa / 64, Ba * NPa), 256, 0, stream>>>(v, vt);
    DotProductAttention_fa_kernel<<<grid, block, 0, stream>>>(q, kh, vt, out);
  } else {
    fa_f32_fallback_kernel<<<grid, block, 0, stream>>>(q, k, v, out);
  }
}
